// SetPooler_10230612099249
// MI455X (gfx1250) — compile-verified, matrix operand order checked by reading, not run on hardware
//
#include <hip/hip_runtime.h>
#include <hip/hip_bf16.h>

typedef __attribute__((ext_vector_type(16))) _Float16 v16h;
typedef __attribute__((ext_vector_type(8)))  _Float16 v8h;
typedef __attribute__((ext_vector_type(8)))  float    v8f;
typedef __attribute__((ext_vector_type(4)))  float    v4f;
typedef __attribute__((ext_vector_type(4)))  int      v4i;

// --------------------------------------------------------------------------
// Optional CDNA5 async global->LDS path (guarded; falls back to sync copies)
// Builtin signature (from probe): (v4i AS1*, v4i AS3*, i32 offset, i32 cpol)
// --------------------------------------------------------------------------
#if defined(__has_builtin)
#  if __has_builtin(__builtin_amdgcn_global_load_async_to_lds_b128)
#    define HAVE_ASYNC_LDS 1
#  endif
#endif
#ifndef HAVE_ASYNC_LDS
#  define HAVE_ASYNC_LDS 0
#endif

#if HAVE_ASYNC_LDS
typedef __attribute__((address_space(1))) v4i g_v4i;
typedef __attribute__((address_space(3))) v4i l_v4i;
#endif

__device__ __forceinline__ void async_copy_b128(const _Float16* g, _Float16* l) {
#if HAVE_ASYNC_LDS
  __builtin_amdgcn_global_load_async_to_lds_b128((g_v4i*)g, (l_v4i*)l, 0, 0);
#else
  *(v8h*)l = *(const v8h*)g;
#endif
}

__device__ __forceinline__ void wait_async_all() {
#if HAVE_ASYNC_LDS
  asm volatile("s_wait_asynccnt 0x0" ::: "memory");
#endif
}

__device__ __forceinline__ v16h make_frag(v8h lo, v8h hi) {
  v16h r;
#pragma unroll
  for (int i = 0; i < 8; ++i) { r[i] = lo[i]; r[i + 8] = hi[i]; }
  return r;
}

__device__ __forceinline__ v8f wmma_f16(v16h a, v16h b, v8f c) {
  // D = A(16x32 f16) x B(32x16 f16) + C(16x16 f32)
  return __builtin_amdgcn_wmma_f32_16x16x32_f16(false, a, false, b, (short)0, c,
                                                false, false);
}

// ---------------------------------------------------------------------------
// Elementwise: X[b,g,n,d] = GA[g,n] * rep_R[b,g,n,d], f32 -> f16
// ---------------------------------------------------------------------------
__global__ __launch_bounds__(256)
void scale_convert_kernel(const float* __restrict__ R,
                          const float* __restrict__ GA,
                          _Float16* __restrict__ X) {
  size_t idx = (size_t)blockIdx.x * blockDim.x + threadIdx.x;
  size_t e0 = idx * 8;
  size_t nd = e0 >> 8;            // ((b*16+g)*512 + n)
  int n = (int)(nd % 512);
  int g = (int)((nd / 512) % 16);
  float ga = GA[g * 512 + n];
  v4f a0 = *(const v4f*)(R + e0);
  v4f a1 = *(const v4f*)(R + e0 + 4);
  v8h o;
#pragma unroll
  for (int j = 0; j < 4; ++j) {
    o[j]     = (_Float16)(a0[j] * ga);
    o[j + 4] = (_Float16)(a1[j] * ga);
  }
  *(v8h*)(X + e0) = o;
}

__global__ void f32_to_f16_kernel(const float* __restrict__ src,
                                  _Float16* __restrict__ dst, int n) {
  int i = blockIdx.x * blockDim.x + threadIdx.x;
  if (i < n) dst[i] = (_Float16)src[i];
}

// ---------------------------------------------------------------------------
// WMMA GEMM: C[M,N] = A[M,K] @ W[K,N] + bias, epilogue by MODE:
//   0: out = acc + bias                    (f16 out)
//   1: out = relu(acc + bias)              (f16 out)
//   2: out = resid + relu(acc + bias)      (f16 out)
//   3: out = acc + bias                    (f32 out)
// Block: 256 threads (8 waves), tile 128x64, BK=32; each wave owns a 32x32
// output (4 WMMA accumulators; 8 ds_load_b128 -> 4 v_wmma per K-step).
// N mult of 64, K mult of 32; M guarded (clamped loads, predicated stores).
// ---------------------------------------------------------------------------
template <int MODE>
__global__ __launch_bounds__(256)
void gemm_bias_kernel(const _Float16* __restrict__ A,
                      const _Float16* __restrict__ W,
                      const float* __restrict__ bias,
                      const _Float16* __restrict__ resid,
                      _Float16* __restrict__ outH,
                      float* __restrict__ outF,
                      int M, int N, int K) {
  __shared__ _Float16 As[128][32];  // (m, k) row-major, 8 KB
  __shared__ _Float16 Bt[64][32];   // (n, k) transposed weights, 4 KB

  const int tid  = threadIdx.x;
  const int lane = tid & 31;
  const int wave = tid >> 5;
  const int wm   = wave & 3;        // 4 row tiles of 32
  const int wn   = wave >> 2;       // 2 col tiles of 32
  const int rowBase = blockIdx.y * 128;
  const int colBase = blockIdx.x * 64;

  const int la   = lane & 15;
  const int koff = (lane >> 4) * 8;

  v8f acc[4] = {v8f{}, v8f{}, v8f{}, v8f{}};

  const int bkr = tid >> 3;                // W staging k 0..31
  const int bnc = (tid & 7) * 8;           // W staging n 0..56

  for (int kb = 0; kb < K; kb += 32) {
    __syncthreads();
    // stage A tile: 128x32 halves = 512 x 16B chunks, 2 per thread (async)
#pragma unroll
    for (int c = 0; c < 2; ++c) {
      int chunk = tid + c * 256;
      int m  = chunk >> 2;
      int k0 = (chunk & 3) * 8;
      int gr = rowBase + m;
      if (gr >= M) gr = M - 1;             // clamp (stores are predicated)
      const _Float16* ap = A + (size_t)gr * K + kb + k0;
      async_copy_b128(ap, &As[m][k0]);
      if (kb + 32 < K) __builtin_prefetch(ap + 32, 0, 1);
    }
    // stage W tile transposed (register transpose, sync)
    {
      const _Float16* wp = W + (size_t)(kb + bkr) * N + colBase + bnc;
      v8h wv = *(const v8h*)wp;
#pragma unroll
      for (int j = 0; j < 8; ++j) Bt[bnc + j][bkr] = wv[j];
    }
    wait_async_all();
    __syncthreads();

    v16h a0 = make_frag(*(const v8h*)(&As[wm * 32 + la][koff]),
                        *(const v8h*)(&As[wm * 32 + la][koff + 16]));
    v16h a1 = make_frag(*(const v8h*)(&As[wm * 32 + 16 + la][koff]),
                        *(const v8h*)(&As[wm * 32 + 16 + la][koff + 16]));
    v16h b0 = make_frag(*(const v8h*)(&Bt[wn * 32 + la][koff]),
                        *(const v8h*)(&Bt[wn * 32 + la][koff + 16]));
    v16h b1 = make_frag(*(const v8h*)(&Bt[wn * 32 + 16 + la][koff]),
                        *(const v8h*)(&Bt[wn * 32 + 16 + la][koff + 16]));
    acc[0] = wmma_f16(a0, b0, acc[0]);
    acc[1] = wmma_f16(a0, b1, acc[1]);
    acc[2] = wmma_f16(a1, b0, acc[2]);
    acc[3] = wmma_f16(a1, b1, acc[3]);
  }

  const int mh = (lane >> 4) * 8;
#pragma unroll
  for (int s = 0; s < 2; ++s) {
#pragma unroll
    for (int r = 0; r < 8; ++r) {
      int row = rowBase + wm * 32 + s * 16 + r + mh;
      if (row < M) {
        int c0 = colBase + wn * 32 + la;
        int c1 = c0 + 16;
        float v0 = acc[s * 2 + 0][r] + bias[c0];
        float v1 = acc[s * 2 + 1][r] + bias[c1];
        if (MODE == 1 || MODE == 2) { v0 = fmaxf(v0, 0.f); v1 = fmaxf(v1, 0.f); }
        if (MODE == 2) {
          v0 += (float)resid[(size_t)row * N + c0];
          v1 += (float)resid[(size_t)row * N + c1];
        }
        if (MODE == 3) {
          outF[(size_t)row * N + c0] = v0;
          outF[(size_t)row * N + c1] = v1;
        } else {
          outH[(size_t)row * N + c0] = (_Float16)v0;
          outH[(size_t)row * N + c1] = (_Float16)v1;
        }
      }
    }
  }
}

// ---------------------------------------------------------------------------
// SAB flash attention: per block = 128 query rows of one (bg, head).
// N=512 keys, D=256, H=4, dh=64. O = q + softmax(q k^T / 16) v
// grid = (4, BG*4); 256 threads = 8 waves x 16 query rows.
// ---------------------------------------------------------------------------
__global__ __launch_bounds__(256)
void sab_attention_kernel(const _Float16* __restrict__ Q,
                          const _Float16* __restrict__ Kg,
                          const _Float16* __restrict__ Vg,
                          _Float16* __restrict__ Og) {
  constexpr int N = 512, D = 256, DH = 64;
  __shared__ _Float16 Ks[64][64];       // (key, dh)
  __shared__ _Float16 Vt[64][64];       // (dh, key) transposed
  __shared__ _Float16 Ps[8][16][64];    // per-wave P tile

  const int tid  = threadIdx.x;
  const int lane = tid & 31;
  const int wave = tid >> 5;
  const int bg   = blockIdx.y >> 2;
  const int h    = blockIdx.y & 3;
  const size_t base = (size_t)bg * N * D;
  const int hoff = h * DH;
  const int qRow = blockIdx.x * 128 + wave * 16;

  const int la   = lane & 15;
  const int koff = (lane >> 4) * 8;
  const int mh   = (lane >> 4) * 8;

  // preload q fragments, prescaled by 1/sqrt(256) = 1/16
  v16h qf[2];
#pragma unroll
  for (int ks = 0; ks < 2; ++ks) {
    const _Float16* qp = Q + base + (size_t)(qRow + la) * D + hoff + ks * 32;
    v8h lo = *(const v8h*)(qp + koff);
    v8h hi = *(const v8h*)(qp + koff + 16);
    v16h f;
#pragma unroll
    for (int i = 0; i < 8; ++i) {
      f[i]     = (_Float16)((float)lo[i] * 0.0625f);
      f[i + 8] = (_Float16)((float)hi[i] * 0.0625f);
    }
    qf[ks] = f;
  }

  v8f acc[4] = {v8f{}, v8f{}, v8f{}, v8f{}};
  float mrow[8], lrow[8];
#pragma unroll
  for (int r = 0; r < 8; ++r) { mrow[r] = -1e30f; lrow[r] = 0.f; }

  for (int kt = 0; kt < N / 64; ++kt) {
    __syncthreads();
#pragma unroll
    for (int c = 0; c < 2; ++c) {
      int chunk = tid * 2 + c;            // 0..511 chunks of 8 halves
      int key = chunk >> 3;
      int d0  = (chunk & 7) * 8;
      const size_t goff = base + (size_t)(kt * 64 + key) * D + hoff + d0;
      async_copy_b128(Kg + goff, &Ks[key][d0]);     // K tile: direct async copy
      v8h vv = *(const v8h*)(Vg + goff);            // V tile: transposed store
#pragma unroll
      for (int j = 0; j < 8; ++j) Vt[d0 + j][key] = vv[j];
    }
    wait_async_all();
    __syncthreads();

    // S = (q/16) @ K^T : 16 x 64 tile
    v8f sacc[4] = {v8f{}, v8f{}, v8f{}, v8f{}};
#pragma unroll
    for (int ks = 0; ks < 2; ++ks) {
#pragma unroll
      for (int t = 0; t < 4; ++t) {
        v16h bf = make_frag(*(const v8h*)(&Ks[t * 16 + la][ks * 32 + koff]),
                            *(const v8h*)(&Ks[t * 16 + la][ks * 32 + koff + 16]));
        sacc[t] = wmma_f16(qf[ks], bf, sacc[t]);
      }
    }

    // online softmax update (rows live in 16-lane halves -> xor reductions)
#pragma unroll
    for (int r = 0; r < 8; ++r) {
      float mt = fmaxf(fmaxf(sacc[0][r], sacc[1][r]),
                       fmaxf(sacc[2][r], sacc[3][r]));
#pragma unroll
      for (int off = 1; off < 16; off <<= 1)
        mt = fmaxf(mt, __shfl_xor(mt, off, 32));
      float mnew  = fmaxf(mrow[r], mt);
      float alpha = __expf(mrow[r] - mnew);
      float rs = 0.f;
#pragma unroll
      for (int t = 0; t < 4; ++t) {
        float p = __expf(sacc[t][r] - mnew);
        rs += p;
        Ps[wave][r + mh][t * 16 + la] = (_Float16)p;
      }
#pragma unroll
      for (int off = 1; off < 16; off <<= 1)
        rs += __shfl_xor(rs, off, 32);
      lrow[r] = lrow[r] * alpha + rs;
      mrow[r] = mnew;
#pragma unroll
      for (int d = 0; d < 4; ++d) acc[d][r] *= alpha;
    }

    // O += P @ V
#pragma unroll
    for (int ks = 0; ks < 2; ++ks) {
      v16h pf = make_frag(*(const v8h*)(&Ps[wave][la][ks * 32 + koff]),
                          *(const v8h*)(&Ps[wave][la][ks * 32 + koff + 16]));
#pragma unroll
      for (int d = 0; d < 4; ++d) {
        v16h vf = make_frag(*(const v8h*)(&Vt[d * 16 + la][ks * 32 + koff]),
                            *(const v8h*)(&Vt[d * 16 + la][ks * 32 + koff + 16]));
        acc[d] = wmma_f16(pf, vf, acc[d]);
      }
    }
  }

  // O = acc/l + q   (residual is the head-projected q)
#pragma unroll
  for (int r = 0; r < 8; ++r) {
    int row = qRow + r + mh;
    float inv = 1.f / lrow[r];
#pragma unroll
    for (int d = 0; d < 4; ++d) {
      size_t idx = base + (size_t)row * D + hoff + d * 16 + la;
      Og[idx] = (_Float16)(acc[d][r] * inv + (float)Q[idx]);
    }
  }
}

// ---------------------------------------------------------------------------
// PMA shared query: qp[256] = pma_S @ pma_wq + pma_bq  (one block, 256 thr)
// ---------------------------------------------------------------------------
__global__ __launch_bounds__(256)
void pma_q_kernel(const float* __restrict__ S, const float* __restrict__ W,
                  const float* __restrict__ b, _Float16* __restrict__ qp) {
  int j = threadIdx.x;
  float acc = b[j];
  for (int i = 0; i < 256; ++i) acc += S[i] * W[i * 256 + j];
  qp[j] = (_Float16)acc;
}

// ---------------------------------------------------------------------------
// PMA 1-query attention: grid (BG=64, H=2), 128 threads. dh=128.
// ---------------------------------------------------------------------------
__global__ __launch_bounds__(128)
void pma_attention_kernel(const _Float16* __restrict__ qp,
                          const _Float16* __restrict__ Kg,
                          const _Float16* __restrict__ Vg,
                          _Float16* __restrict__ Og) {
  constexpr int N = 512, D = 256, DH = 128;
  __shared__ float p[N];
  __shared__ float red[128];
  const int t  = threadIdx.x;
  const int bg = blockIdx.x;
  const int h  = blockIdx.y;
  const size_t base = (size_t)bg * N * D;
  const int hoff = h * DH;

  float lmax = -1e30f;
  for (int k = t; k < N; k += 128) {
    const _Float16* kr = Kg + base + (size_t)k * D + hoff;
    float s = 0.f;
    for (int j = 0; j < DH; ++j) s += (float)qp[hoff + j] * (float)kr[j];
    s *= 0.0625f;
    p[k] = s;
    lmax = fmaxf(lmax, s);
  }
  red[t] = lmax; __syncthreads();
  for (int o = 64; o > 0; o >>= 1) {
    if (t < o) red[t] = fmaxf(red[t], red[t + o]);
    __syncthreads();
  }
  float m = red[0];
  __syncthreads();
  float lsum = 0.f;
  for (int k = t; k < N; k += 128) {
    float e = __expf(p[k] - m);
    p[k] = e;
    lsum += e;
  }
  red[t] = lsum; __syncthreads();
  for (int o = 64; o > 0; o >>= 1) {
    if (t < o) red[t] += red[t + o];
    __syncthreads();
  }
  float inv = 1.f / red[0];
  __syncthreads();

  float acc = 0.f;
  for (int k = 0; k < N; ++k)
    acc += p[k] * (float)Vg[base + (size_t)k * D + hoff + t];
  Og[(size_t)bg * D + hoff + t] = (_Float16)((float)qp[hoff + t] + acc * inv);
}

// ---------------------------------------------------------------------------
// Host orchestration
// ---------------------------------------------------------------------------
extern "C" void kernel_launch(void* const* d_in, const int* in_sizes, int n_in,
                              void* d_out, int out_size, void* d_ws, size_t ws_size,
                              hipStream_t stream) {
  (void)in_sizes; (void)n_in; (void)out_size; (void)ws_size;
  const int B = 4, G = 16, N = 512, D = 256;
  const int BG = B * G;                 // 64
  const int M_big = BG * N;             // 32768

  char* ws = (char*)d_ws;
  _Float16* Xh  = (_Float16*)(ws);                            // 16 MB
  _Float16* Bq  = (_Float16*)(ws + (size_t)16 * (1 << 20));
  _Float16* Bk  = (_Float16*)(ws + (size_t)32 * (1 << 20));
  _Float16* Bv  = (_Float16*)(ws + (size_t)48 * (1 << 20));
  _Float16* Bo  = (_Float16*)(ws + (size_t)64 * (1 << 20));
  _Float16* WH  = (_Float16*)(ws + (size_t)80 * (1 << 20));   // f16 weights
  char* smallb  = ws + (size_t)84 * (1 << 20);
  _Float16* qp   = (_Float16*)(smallb);
  _Float16* pmaO = (_Float16*)(smallb + (1 << 16));
  _Float16* pmaY = (_Float16*)(smallb + 2 * (1 << 16));
  _Float16* h1   = (_Float16*)(smallb + 3 * (1 << 16));
  _Float16* h2   = (_Float16*)(smallb + 4 * (1 << 16));

  // 1. X = GA * rep_R, f32 -> f16
  scale_convert_kernel<<<4096, 256, 0, stream>>>(
      (const float*)d_in[0], (const float*)d_in[1], Xh);

  // 2. weight conversions to f16
  struct WC { int idx; size_t off; int n; };
  const WC wc[15] = {
      {2, 0, 65536},       {4, 65536, 65536},   {6, 131072, 65536},
      {8, 196608, 65536},  {10, 262144, 65536}, {12, 327680, 65536},
      {14, 393216, 65536}, {16, 458752, 65536}, {19, 524288, 65536},
      {21, 589824, 65536}, {23, 655360, 65536}, {25, 720896, 65536},
      {27, 786432, 131072},{29, 917504, 262144},{31, 1179648, 131072}};
  for (int i = 0; i < 15; ++i)
    f32_to_f16_kernel<<<(wc[i].n + 255) / 256, 256, 0, stream>>>(
        (const float*)d_in[wc[i].idx], WH + wc[i].off, wc[i].n);

  const dim3 gBig(D / 64, M_big / 128);       // (4, 256)
  const dim3 gSmall(D / 64, 1);               // M=64 rows fit one tile row

  // 3. two SAB layers (X lives in Xh, updated in place via Bo)
  for (int s = 0; s < 2; ++s) {
    const size_t wb = (size_t)s * 262144;     // f16 weight base for this SAB
    const int ib = 2 + s * 8;                 // input index base
    gemm_bias_kernel<0><<<gBig, 256, 0, stream>>>(
        Xh, WH + wb + 0,      (const float*)d_in[ib + 1], nullptr, Bq, nullptr,
        M_big, D, D);
    gemm_bias_kernel<0><<<gBig, 256, 0, stream>>>(
        Xh, WH + wb + 65536,  (const float*)d_in[ib + 3], nullptr, Bk, nullptr,
        M_big, D, D);
    gemm_bias_kernel<0><<<gBig, 256, 0, stream>>>(
        Xh, WH + wb + 131072, (const float*)d_in[ib + 5], nullptr, Bv, nullptr,
        M_big, D, D);
    sab_attention_kernel<<<dim3(N / 128, BG * 4), 256, 0, stream>>>(Bq, Bk, Bv, Bo);
    // X = O + relu(O @ wo + bo)
    gemm_bias_kernel<2><<<gBig, 256, 0, stream>>>(
        Bo, WH + wb + 196608, (const float*)d_in[ib + 7], Bo, Xh, nullptr,
        M_big, D, D);
  }

  // 4. PMA: k/v projections of X, shared query, 1-query attention
  {
    gemm_bias_kernel<0><<<gBig, 256, 0, stream>>>(
        Xh, WH + 589824, (const float*)d_in[22], nullptr, Bk, nullptr, M_big, D, D);
    gemm_bias_kernel<0><<<gBig, 256, 0, stream>>>(
        Xh, WH + 655360, (const float*)d_in[24], nullptr, Bv, nullptr, M_big, D, D);
    pma_q_kernel<<<1, 256, 0, stream>>>(
        (const float*)d_in[18], (const float*)d_in[19], (const float*)d_in[20], qp);
    pma_attention_kernel<<<dim3(BG, 2), 128, 0, stream>>>(qp, Bk, Bv, pmaO);
    // Y = O + relu(O @ pma_wo + pma_bo)   (M=64)
    gemm_bias_kernel<2><<<gSmall, 256, 0, stream>>>(
        pmaO, WH + 720896, (const float*)d_in[26], pmaO, pmaY, nullptr, BG, D, D);
  }

  // 5. MLP: 256 -> 512 -> 512 -> 256 (final f32 to d_out)
  gemm_bias_kernel<1><<<dim3(512 / 64, 1), 256, 0, stream>>>(
      pmaY, WH + 786432, (const float*)d_in[28], nullptr, h1, nullptr, BG, 512, 256);
  gemm_bias_kernel<1><<<dim3(512 / 64, 1), 256, 0, stream>>>(
      h1, WH + 917504, (const float*)d_in[30], nullptr, h2, nullptr, BG, 512, 512);
  gemm_bias_kernel<3><<<dim3(256 / 64, 1), 256, 0, stream>>>(
      h2, WH + 1179648, (const float*)d_in[32], nullptr, nullptr, (float*)d_out,
      BG, 256, 512);
}
